// WarpedNonlinearity_67156108640846
// MI455X (gfx1250) — matrix operand order repaired and checked by reading, not compiled, not run on hardware
//
#include <hip/hip_runtime.h>
#include <math.h>

// ---------------------------------------------------------------------------
// WarpedNonlinearity on MI455X (gfx1250, wave32, WMMA f16 -> f32 accumulate)
//
// Algebraic reduction of the FFT pipeline (per 64x64 image):
//     y = 0.25 * A^T * silu( A * x * A^T ) * A
// where A (128x64) is the circular-sinc 2x interpolation matrix with the
// Nyquist bin half-split (the LPF(1/2)+stride-2 operator equals A^T/2).
// One workgroup per image; 4 GEMMs done with v_wmma_f32_16x16x32_f16.
// ---------------------------------------------------------------------------

typedef __attribute__((ext_vector_type(16))) _Float16 v16h;
typedef __attribute__((ext_vector_type(8)))  _Float16 v8h;
typedef __attribute__((ext_vector_type(8)))  float    v8f;

#define SX 72    // padded LDS row stride (halves) for 64-wide rows  (144 B)
#define SW 136   // padded LDS row stride (halves) for 128-wide rows (272 B)

// LDS carve offsets (in halves)
#define OFF_XT   0                       // x^T          [64][SX]
#define OFF_ARM  (OFF_XT   + 64  * SX)   // A row-major  [128][SX]
#define OFF_ATRM (OFF_ARM  + 128 * SX)   // A^T row-major[64][SW]
#define OFF_T1   (OFF_ATRM + 64  * SW)   // T1 = A*x     [128][SX] (row-major)
#define OFF_ACM  (OFF_T1   + 128 * SX)   // silu(u) col-major [128][SW]
#define OFF_T2   (OFF_ACM  + 128 * SW)   // T2 = A^T*a   [64][SW]  (row-major)
#define SMEM_HALVES (OFF_T2 + 64 * SW)   // = 57856 halves = 115712 B

// --------------------------- A-matrix generator -----------------------------
// A[m,n] = (1/64)[1 + 2*sum_{k=1..31} cos(2pi k (m/2-n)/64) + cos(pi(m/2-n))]
// Even m rows are exact Kronecker deltas (sinc interpolation is interpolatory).
__global__ void build_sinc_matrix(float* __restrict__ Aw) {
    int idx = blockIdx.x * blockDim.x + threadIdx.x;
    if (idx >= 128 * 64) return;
    int m = idx >> 6, n = idx & 63;
    float val;
    if ((m & 1) == 0) {
        val = ((m >> 1) == n) ? 1.0f : 0.0f;
    } else {
        const float PI = 3.14159265358979323846f;
        float t = 0.5f * (float)m - (float)n;
        float s = 1.0f + cosf(PI * t);
        for (int k = 1; k <= 31; ++k)
            s += 2.0f * cosf((PI / 32.0f) * (float)k * t);
        val = s * (1.0f / 64.0f);
    }
    Aw[idx] = val;
}

// ------------------------- WMMA operand loaders -----------------------------
// Left (A) operand, 16x32 f16, left matrix stored row-major in LDS.
// Lane (r = lane&15, hf = lane>>4): row = rowBase+r,
//   elems 0..7  <-> K = k0 + hf*8 + 0..7      (VGPR0-3)
//   elems 8..15 <-> K = k0 + 16 + hf*8 + 0..7 (VGPR4-7)
__device__ __forceinline__ v16h load_lhs(const _Float16* Lrm, int stride,
                                         int rowBase, int k0, int r, int hf) {
    const _Float16* p = Lrm + (rowBase + r) * stride + k0 + hf * 8;
    v8h lo = *(const v8h*)p;          // ds_load_b128
    v8h hi = *(const v8h*)(p + 16);   // ds_load_b128
    v16h o;
#pragma unroll
    for (int i = 0; i < 8; ++i) { o[i] = lo[i]; o[i + 8] = hi[i]; }
    return o;
}

// Right (B) operand, 32x16 f16. Right matrix R stored as a column-major
// buffer Rcm[col][k] so each lane reads two runs of 8 contiguous halves.
// Lane: col = colBase + r, K = k0 + hf*8 + 0..7, k0 + 16 + hf*8 + 0..7.
__device__ __forceinline__ v16h load_rhs(const _Float16* Rcm, int stride,
                                         int colBase, int k0, int r, int hf) {
    const _Float16* p = Rcm + (colBase + r) * stride + k0 + hf * 8;
    v8h lo = *(const v8h*)p;          // ds_load_b128
    v8h hi = *(const v8h*)(p + 16);   // ds_load_b128
    v16h o;
#pragma unroll
    for (int i = 0; i < 8; ++i) { o[i] = lo[i]; o[i + 8] = hi[i]; }
    return o;
}

// --------------------------------- main -------------------------------------
__global__ __launch_bounds__(256) void warped_nl_kernel(
        const float* __restrict__ x, const float* __restrict__ Aw,
        float* __restrict__ out) {
    extern __shared__ _Float16 sm[];
    _Float16* xT   = sm + OFF_XT;    // x^T            [64][SX]
    _Float16* Arm  = sm + OFF_ARM;   // A  row-major   [128][SX]
    _Float16* ATrm = sm + OFF_ATRM;  // A^T row-major  [64][SW]
    _Float16* T1   = sm + OFF_T1;    // T1 row-major   [128][SX]
    _Float16* Acm  = sm + OFF_ACM;   // a = silu(u), column-major [128][SW]
    _Float16* T2   = sm + OFF_T2;    // T2 row-major   [64][SW]

    const int tid  = threadIdx.x;
    const int lane = tid & 31;
    const int w    = tid >> 5;       // wave id 0..7
    const int r    = lane & 15;
    const int hf   = lane >> 4;

    // ---- stage A (f32 -> f16) into both layouts ----
    for (int i = tid; i < 128 * 64; i += 256) {
        int m = i >> 6, n = i & 63;
        _Float16 v = (_Float16)Aw[i];
        Arm[m * SX + n]  = v;
        ATrm[n * SW + m] = v;
    }
    // ---- stage x transposed (f32 -> f16), coalesced float4 global loads ----
    const float* xin = x + (size_t)blockIdx.x * 4096;
    for (int c4 = 0; c4 < 4; ++c4) {
        int base = c4 * 1024 + tid * 4;
        const float4 v = *(const float4*)(xin + base);
        int n = base >> 6, j = base & 63;       // x[n][j..j+3]
        xT[(j + 0) * SX + n] = (_Float16)v.x;
        xT[(j + 1) * SX + n] = (_Float16)v.y;
        xT[(j + 2) * SX + n] = (_Float16)v.z;
        xT[(j + 3) * SX + n] = (_Float16)v.w;
    }
    __syncthreads();

    // ---- Step 1: T1^T = x^T (64x64) * A^T (64x128), K=64.
    // Left rows: xT. Right cm-buffer for A^T is Arm. Storing D column-major
    // yields exactly T1 row-major: one b128 store per lane.
    {
        int tr = w & 3;                           // j-tile (rows of T1^T)
        for (int q = 0; q < 4; ++q) {
            int tc = (w >> 2) * 4 + q;            // m-tile (cols of T1^T)
            v8f c = {};
#pragma unroll
            for (int kk = 0; kk < 2; ++kk) {
                v16h a = load_lhs(xT,  SX, tr * 16, kk * 32, r, hf);
                v16h b = load_rhs(Arm, SX, tc * 16, kk * 32, r, hf);
                c = __builtin_amdgcn_wmma_f32_16x16x32_f16(
                        false, a, false, b, (short)0, c, false, false);
            }
            v8h h;
#pragma unroll
            for (int i = 0; i < 8; ++i) h[i] = (_Float16)c[i];
            *(v8h*)&T1[(tc * 16 + r) * SX + tr * 16 + hf * 8] = h;
        }
    }
    __syncthreads();

    // ---- Step 2: u = T1 (128x64) * A^T (64x128), K=64; silu; store a
    // column-major (one b128 store per lane, ready as Step-3 left operand).
    {
        int tr = w;                               // m-tile
        for (int tc = 0; tc < 8; ++tc) {          // p-tile
            v8f c = {};
#pragma unroll
            for (int kk = 0; kk < 2; ++kk) {
                v16h a = load_lhs(T1,  SX, tr * 16, kk * 32, r, hf);
                v16h b = load_rhs(Arm, SX, tc * 16, kk * 32, r, hf);
                c = __builtin_amdgcn_wmma_f32_16x16x32_f16(
                        false, a, false, b, (short)0, c, false, false);
            }
            v8h h;
#pragma unroll
            for (int i = 0; i < 8; ++i) {
                float s = c[i];
                s = s / (1.0f + __expf(-s));      // silu in fp32
                h[i] = (_Float16)s;
            }
            *(v8h*)&Acm[(tc * 16 + r) * SW + tr * 16 + hf * 8] = h;
        }
    }
    __syncthreads();

    // ---- Step 3: T2^T = a^T (128x128) * A (128x64), K=128.
    // Left rows: Acm (= a^T row-major). Right cm-buffer for A is ATrm.
    // Column-major D store == T2 row-major.
    {
        int tr = w;                               // p-tile
        for (int tc = 0; tc < 4; ++tc) {          // i-tile
            v8f c = {};
#pragma unroll
            for (int kk = 0; kk < 4; ++kk) {
                v16h a = load_lhs(Acm,  SW, tr * 16, kk * 32, r, hf);
                v16h b = load_rhs(ATrm, SW, tc * 16, kk * 32, r, hf);
                c = __builtin_amdgcn_wmma_f32_16x16x32_f16(
                        false, a, false, b, (short)0, c, false, false);
            }
            v8h h;
#pragma unroll
            for (int i = 0; i < 8; ++i) h[i] = (_Float16)c[i];
            *(v8h*)&T2[(tc * 16 + r) * SW + tr * 16 + hf * 8] = h;
        }
    }
    __syncthreads();

    // ---- Step 4: y = 0.25 * T2 (64x128) * A (128x64), K=128 -> global f32.
    {
        float* yo = out + (size_t)blockIdx.x * 4096;
        int tr = w >> 1;                          // i-tile
        for (int q = 0; q < 2; ++q) {
            int tc = (w & 1) * 2 + q;             // q-tile
            v8f c = {};
#pragma unroll
            for (int kk = 0; kk < 4; ++kk) {
                v16h a = load_lhs(T2,   SW, tr * 16, kk * 32, r, hf);
                v16h b = load_rhs(ATrm, SW, tc * 16, kk * 32, r, hf);
                c = __builtin_amdgcn_wmma_f32_16x16x32_f16(
                        false, a, false, b, (short)0, c, false, false);
            }
#pragma unroll
            for (int i = 0; i < 8; ++i) {
                int row = tr * 16 + hf * 8 + i;   // C layout: M = hf*8+i
                int col = tc * 16 + r;            //           N = r
                yo[row * 64 + col] = 0.25f * c[i];
            }
        }
    }
}

// ------------------------------- launcher -----------------------------------
extern "C" void kernel_launch(void* const* d_in, const int* in_sizes, int n_in,
                              void* d_out, int out_size, void* d_ws, size_t ws_size,
                              hipStream_t stream) {
    (void)n_in; (void)out_size; (void)ws_size;
    const float* x  = (const float*)d_in[0];
    float*       y  = (float*)d_out;
    float*       Aw = (float*)d_ws;                 // 128*64 f32 = 32 KB

    const int images = in_sizes[0] / (64 * 64);     // B*C = 2048

    build_sinc_matrix<<<(128 * 64 + 255) / 256, 256, 0, stream>>>(Aw);

    const size_t smem = (size_t)SMEM_HALVES * sizeof(_Float16);  // 115712 B
    warped_nl_kernel<<<images, 256, smem, stream>>>(x, Aw, y);
}
